// Decoder_recon_28321014350221
// MI455X (gfx1250) — matrix operand order repaired and checked by reading, not compiled, not run on hardware
//
#include <hip/hip_runtime.h>
#include <math.h>

typedef __attribute__((ext_vector_type(16))) _Float16 v16h;
typedef __attribute__((ext_vector_type(8)))  _Float16 v8h;
typedef __attribute__((ext_vector_type(8)))  float    v8f;

#define E_DIM   512
#define NHEAD   8
#define HD      64
#define DFF_DIM 2048
#define NAG     32
#define SNUM    8
#define LQ      384
#define LKMEM   256
#define MQ      (LQ * SNUM)     /* 3072 q tokens  */
#define MK      (LKMEM * SNUM)  /* 2048 kv tokens */
#define NEGINF  (-1e9f)
#define LDT     40              /* padded LDS row stride (halves) for 32-K tiles  */
#define LDQ72   72              /* padded LDS row stride (halves) for 64-K tiles  */

// K-permutation so an A-fragment (ISA 7.12.2: lanes0-15 K={0..7,16..23},
// lanes16-31 K={8..15,24..31}) becomes 16 contiguous halves per lane.
__device__ __forceinline__ int kperm(int K) {
  return ((K & 8) << 1) + ((K & 16) >> 1) + (K & 7);
}

__device__ __forceinline__ uint2 pack4h(float4 f) {
  union { _Float16 h[4]; uint2 u; } u;
  u.h[0] = (_Float16)f.x; u.h[1] = (_Float16)f.y;
  u.h[2] = (_Float16)f.z; u.h[3] = (_Float16)f.w;
  return u.u;
}

__device__ __forceinline__ v16h cat8(v8h lo, v8h hi) {
  return __builtin_shufflevector(lo, hi, 0,1,2,3,4,5,6,7,8,9,10,11,12,13,14,15);
}

// ======================= generic WMMA GEMM =======================
// out[M,N] = act(A[M,K] @ W[N,K]^T + bias[N]) (+ res[M,N])
// A always f16; W f32 (harness weights). MODE: 0 = write f32 Cf,
// 1 = write f16 Ch, 2 = write both.
// M%128==0, N%128==0, K%32==0. 256 threads = 8 waves, 128x128 tile,
// wave = 32x64 -> 8 WMMAs / 32-K step. Double-buffered LDS pipeline:
// one barrier per K-step (stage k+1 while computing k, prefetch k+2).
template<int ACT, int RES, int MODE>
__global__ __launch_bounds__(256)
void gemm_wmma(const _Float16* __restrict__ A, const float* __restrict__ W,
               const float* __restrict__ bias, const float* __restrict__ res,
               float* __restrict__ Cf, _Float16* __restrict__ Ch, int M, int N, int K)
{
  __shared__ _Float16 sA[2][128 * LDT];   // K-permuted
  __shared__ _Float16 sB[2][128 * LDT];   // K-permuted (same as sA)
  const int n0   = blockIdx.x * 128;
  const int m0   = blockIdx.y * 128;
  const int tid  = threadIdx.x;
  const int wave = tid >> 5;
  const int lane = tid & 31;
  const int wr   = wave & 3;           // m sub-tile (32 rows)
  const int wc   = wave >> 2;          // n sub-tile (64 cols)
  const int fr   = lane & 15;
  const int fo   = (lane >> 4) * 16;   // contiguous fragment offset (halves)
  (void)M;

  v8h    rah[2];                       // A prefetch regs (pure f16 copy)
  float4 rw[4];                        // W prefetch regs (f32 -> cvt once)

  auto fetch = [&](int k0) {
#pragma unroll
    for (int it = 0; it < 2; ++it) {   // 128 rows x 4 v8h = 512 / 256 thr
      int idx = tid + it * 256;
      int r = idx >> 2, cb = (idx & 3) * 8;
      rah[it] = *(const v8h*)(A + (size_t)(m0 + r) * K + k0 + cb);
    }
#pragma unroll
    for (int it = 0; it < 4; ++it) {   // 128 rows x 8 float4 = 1024 / 256 thr
      int idx = tid + it * 256;
      int r = idx >> 3, cb = (idx & 7) * 4;
      rw[it] = *(const float4*)(W + (size_t)(n0 + r) * K + k0 + cb);
    }
  };
  auto stage = [&](int buf) {
#pragma unroll
    for (int it = 0; it < 2; ++it) {
      int idx = tid + it * 256;
      int r = idx >> 2, cb = (idx & 3) * 8;   // 8-half chunk stays inside one kperm block
      *(v8h*)&sA[buf][r * LDT + kperm(cb)] = rah[it];
    }
#pragma unroll
    for (int it = 0; it < 4; ++it) {
      int idx = tid + it * 256;
      int r = idx >> 3, cb = (idx & 7) * 4;
      *(uint2*)&sB[buf][r * LDT + kperm(cb)] = pack4h(rw[it]);
    }
  };

  v8f acc[2][4];
#pragma unroll
  for (int i = 0; i < 2; ++i)
#pragma unroll
    for (int nt = 0; nt < 4; ++nt)
      acc[i][nt] = (v8f){0.f,0.f,0.f,0.f,0.f,0.f,0.f,0.f};

  const int KT = K >> 5;
  fetch(0);
  stage(0);
  if (KT > 1) fetch(32);

  for (int kt = 0; kt < KT; ++kt) {
    __syncthreads();
    if (kt + 1 < KT) stage((kt + 1) & 1);
    if (kt + 2 < KT) fetch((kt + 2) << 5);
    const _Float16* cA = sA[kt & 1];
    const _Float16* cB = sB[kt & 1];
    v16h af[2], bf[4];
#pragma unroll
    for (int i = 0; i < 2; ++i) {
      int base = (wr * 32 + i * 16 + fr) * LDT + fo;
      af[i] = cat8(*(const v8h*)&cA[base], *(const v8h*)&cA[base + 8]);
    }
#pragma unroll
    for (int nt = 0; nt < 4; ++nt) {
      int base = (wc * 64 + nt * 16 + fr) * LDT + fo;
      bf[nt] = cat8(*(const v8h*)&cB[base], *(const v8h*)&cB[base + 8]);
    }
#pragma unroll
    for (int i = 0; i < 2; ++i)
#pragma unroll
      for (int nt = 0; nt < 4; ++nt)
        acc[i][nt] = __builtin_amdgcn_wmma_f32_16x16x32_f16(false, af[i], false, bf[nt],
                                                            (short)0, acc[i][nt], false, false);
  }

#pragma unroll
  for (int i = 0; i < 2; ++i)
#pragma unroll
    for (int nt = 0; nt < 4; ++nt) {
      int n  = n0 + wc * 64 + nt * 16 + fr;
      int mb = m0 + wr * 32 + i * 16 + ((lane >> 4) << 3);
      float bv = bias[n];
#pragma unroll
      for (int r = 0; r < 8; ++r) {
        int m = mb + r;
        float val = acc[i][nt][r] + bv;
        if (ACT) val = fmaxf(val, 0.f);
        if (RES) val += res[(size_t)m * N + n];
        if (MODE != 1) Cf[(size_t)m * N + n] = val;
        if (MODE != 0) Ch[(size_t)m * N + n] = (_Float16)val;
      }
    }
}

// =================== agent-aware attention scores (f16 in) ===================
template<int ARMASK>
__global__ __launch_bounds__(32)
void attn_scores(const _Float16* __restrict__ q,  int ldq,
                 const _Float16* __restrict__ k,  int ldk,
                 const _Float16* __restrict__ qs, int ldqs,
                 const _Float16* __restrict__ ks, int ldks,
                 float* __restrict__ scores, int Lk)
{
  __shared__ _Float16 sQ[16 * LDQ72], sQs[16 * LDQ72];  // K-permuted per 32-block
  __shared__ _Float16 sK[16 * LDQ72], sKs[16 * LDQ72];  // K-permuted per 32-block
  const int i0 = blockIdx.x * 16;
  const int j0 = blockIdx.y * 16;
  const int bh = blockIdx.z;           // 0..63
  const int b  = bh & 7, h = bh >> 3;
  const int lane = threadIdx.x;
  const int coff = h * HD;
  const int fr = lane & 15;
  const int fo = (lane >> 4) * 16;
#pragma unroll
  for (int it = 0; it < 4; ++it) {     // 16 rows x 8 v8h chunks = 128 / 32
    int idx = lane + it * 32;
    int r = idx >> 3, cb = (idx & 7) * 8;
    int qpos = (cb & 32) + kperm(cb & 31);
    *(v8h*)&sQ [r * LDQ72 + qpos] = *(const v8h*)(q  + (size_t)((i0 + r) * SNUM + b) * ldq  + coff + cb);
    *(v8h*)&sQs[r * LDQ72 + qpos] = *(const v8h*)(qs + (size_t)((i0 + r) * SNUM + b) * ldqs + coff + cb);
    *(v8h*)&sK [r * LDQ72 + qpos] = *(const v8h*)(k  + (size_t)((j0 + r) * SNUM + b) * ldk  + coff + cb);
    *(v8h*)&sKs[r * LDQ72 + qpos] = *(const v8h*)(ks + (size_t)((j0 + r) * SNUM + b) * ldks + coff + cb);
  }
  __syncthreads();
  v8f accI = (v8f){0.f,0.f,0.f,0.f,0.f,0.f,0.f,0.f};
  v8f accS = (v8f){0.f,0.f,0.f,0.f,0.f,0.f,0.f,0.f};
#pragma unroll
  for (int kb = 0; kb < HD; kb += 32) {
    int qb = fr * LDQ72 + kb + fo;
    v16h aI = cat8(*(const v8h*)&sQ [qb], *(const v8h*)&sQ [qb + 8]);
    v16h aS = cat8(*(const v8h*)&sQs[qb], *(const v8h*)&sQs[qb + 8]);
    v16h bI = cat8(*(const v8h*)&sK [qb], *(const v8h*)&sK [qb + 8]);
    v16h bS = cat8(*(const v8h*)&sKs[qb], *(const v8h*)&sKs[qb + 8]);
    accI = __builtin_amdgcn_wmma_f32_16x16x32_f16(false, aI, false, bI, (short)0, accI, false, false);
    accS = __builtin_amdgcn_wmma_f32_16x16x32_f16(false, aS, false, bS, (short)0, accS, false, false);
  }
#pragma unroll
  for (int r = 0; r < 8; ++r) {
    int i = i0 + ((lane >> 4) << 3) + r;
    int j = j0 + fr;
    float s = (((i & (NAG - 1)) == (j & (NAG - 1))) ? accS[r] : accI[r]) * 0.125f;
    if (ARMASK && ((i >> 5) < (j >> 5))) s += NEGINF;   // tk > tq masked
    scores[((size_t)bh * LQ + i) * Lk + j] = s;
  }
}

// ========================= row softmax -> f16 probs =========================
__global__ __launch_bounds__(32)
void softmax_rows(const float* __restrict__ scores, _Float16* __restrict__ probs, int Lk)
{
  const int row  = blockIdx.x;
  const int lane = threadIdx.x;
  const float* src = scores + (size_t)row * Lk;
  _Float16*    dst = probs  + (size_t)row * Lk;
  float m = -INFINITY;
  for (int j = lane; j < Lk; j += 32) m = fmaxf(m, src[j]);
#pragma unroll
  for (int o = 16; o; o >>= 1) m = fmaxf(m, __shfl_xor(m, o, 32));
  float vals[12];
  float sum = 0.f;
  int cnt = 0;
  for (int j = lane; j < Lk; j += 32) { float e = __expf(src[j] - m); vals[cnt++] = e; sum += e; }
#pragma unroll
  for (int o = 16; o; o >>= 1) sum += __shfl_xor(sum, o, 32);
  float inv = 1.f / sum;
  cnt = 0;
  for (int j = lane; j < Lk; j += 32) dst[j] = (_Float16)(vals[cnt++] * inv);
}

// ============================= P @ V (WMMA, f16 in/out) =============================
__global__ __launch_bounds__(32)
void attn_pv(const _Float16* __restrict__ probs, const _Float16* __restrict__ v, int ldv,
             _Float16* __restrict__ out, int Lk)
{
  __shared__ _Float16 sVT[16 * LDT];   // transposed: sVT[n][kk]
  const int i0 = blockIdx.x * 16;
  const int d0 = blockIdx.y * 16;
  const int bh = blockIdx.z;
  const int b  = bh & 7, h = bh >> 3;
  const int lane = threadIdx.x;
  const int fr = lane & 15;
  const int fo = (lane >> 4) * 16;
  const int g  = lane >> 1;            // kk pair index 0..15
  const int nb = (lane & 1) * 8;       // n chunk
  v8f acc = (v8f){0.f,0.f,0.f,0.f,0.f,0.f,0.f,0.f};
  const _Float16* prow = probs + ((size_t)bh * LQ + i0 + fr) * Lk;
  for (int k0 = 0; k0 < Lk; k0 += 32) {
    // stage V transposed: lane loads 2 adjacent kk rows, stores packed kk-pairs (b32)
    v8h f0 = *(const v8h*)(v + (size_t)((k0 + 2 * g)     * SNUM + b) * ldv + h * HD + d0 + nb);
    v8h f1 = *(const v8h*)(v + (size_t)((k0 + 2 * g + 1) * SNUM + b) * ldv + h * HD + d0 + nb);
#pragma unroll
    for (int qd = 0; qd < 8; ++qd) {
      union { _Float16 h2[2]; unsigned u; } p;
      p.h2[0] = f0[qd]; p.h2[1] = f1[qd];
      *(unsigned*)&sVT[(nb + qd) * LDT + 2 * g] = p.u;
    }
    __syncthreads();
    // A fragment: two contiguous 16B chunks straight from global f16 probs
    const _Float16* pbase = prow + k0;
    v16h a  = cat8(*(const v8h*)(pbase + ((lane >> 4) << 4)),
                   *(const v8h*)(pbase + 8 + ((lane >> 4) << 4)));
    int vb = fr * LDT + fo;
    v16h bf = cat8(*(const v8h*)&sVT[vb], *(const v8h*)&sVT[vb + 8]);
    acc = __builtin_amdgcn_wmma_f32_16x16x32_f16(false, a, false, bf, (short)0, acc, false, false);
    __syncthreads();
  }
#pragma unroll
  for (int r = 0; r < 8; ++r) {
    int i = i0 + ((lane >> 4) << 3) + r;
    out[(size_t)(i * SNUM + b) * E_DIM + h * HD + d0 + fr] = (_Float16)acc[r];
  }
}

// ==================== LayerNorm (dual f32 + f16 output) ====================
__global__ __launch_bounds__(32)
void layernorm(const float* __restrict__ x, const float* __restrict__ g,
               const float* __restrict__ bt, float* __restrict__ y,
               _Float16* __restrict__ yh)
{
  const int t    = blockIdx.x;
  const int lane = threadIdx.x;
  const float* row = x + (size_t)t * E_DIM;
  float v[16];
  float s = 0.f;
#pragma unroll
  for (int i = 0; i < 16; ++i) { v[i] = row[lane + i * 32]; s += v[i]; }
#pragma unroll
  for (int o = 16; o; o >>= 1) s += __shfl_xor(s, o, 32);
  float mu = s * (1.f / E_DIM);
  float vs = 0.f;
#pragma unroll
  for (int i = 0; i < 16; ++i) { float d = v[i] - mu; vs += d * d; }
#pragma unroll
  for (int o = 16; o; o >>= 1) vs += __shfl_xor(vs, o, 32);
  float rstd = rsqrtf(vs * (1.f / E_DIM) + 1e-5f);
#pragma unroll
  for (int i = 0; i < 16; ++i) {
    int c = lane + i * 32;
    float val = (v[i] - mu) * rstd * g[c] + bt[c];
    y [(size_t)t * E_DIM + c] = val;
    yh[(size_t)t * E_DIM + c] = (_Float16)val;
  }
}

// ===================== small VALU helper kernels =====================
__global__ void pe_kernel(float* __restrict__ pe)
{
  int gid = blockIdx.x * blockDim.x + threadIdx.x;
  if (gid >= 12 * E_DIM) return;
  int p = gid / E_DIM, c = gid % E_DIM;
  float i2  = (float)((c >> 1) << 1);
  float div = __expf(i2 * (-logf(10000.f) / (float)E_DIM));
  float ang = (float)p * div;
  pe[gid] = (c & 1) ? cosf(ang) : sinf(ang);
}

// Xh[t,0:512] = [dec_in,z] @ input_fc_w^T + b ; Xh[t,512:1024] = PE[t/256]  (f16 out)
__global__ void prep_x1024(const float* __restrict__ v, const float* __restrict__ z,
                           const float* __restrict__ wfc, const float* __restrict__ bfc,
                           const float* __restrict__ pe, _Float16* __restrict__ Xh)
{
  int gid = blockIdx.x * blockDim.x + threadIdx.x;
  if (gid >= MQ * E_DIM) return;
  int t = gid / E_DIM, c = gid % E_DIM;
  int row = t / SNUM;
  const float* w  = wfc + (size_t)c * 34;
  const float* zr = z + (size_t)t * 32;
  float acc = bfc[c] + v[row * 2 + 0] * w[0] + v[row * 2 + 1] * w[1];
#pragma unroll
  for (int k = 0; k < 32; ++k) acc += zr[k] * w[2 + k];
  Xh[(size_t)t * 1024 + c]       = (_Float16)acc;
  Xh[(size_t)t * 1024 + 512 + c] = (_Float16)pe[(size_t)(t / 256) * E_DIM + c];
}

__global__ void repeat_venc(const float* __restrict__ venc, _Float16* __restrict__ out)
{
  int gid = blockIdx.x * blockDim.x + threadIdx.x;
  if (gid >= MK * E_DIM) return;
  int u = gid / E_DIM, c = gid % E_DIM;
  out[gid] = (_Float16)venc[(size_t)(u / SNUM) * E_DIM + c];
}

// out[s,l,n,c] = h2[t]·out_fc_w[c] + out_fc_b[c] + v_in[0,0,n,c] + scene[c]
__global__ void head_out(const float* __restrict__ h2, const float* __restrict__ w,
                         const float* __restrict__ bias, const float* __restrict__ vin,
                         const float* __restrict__ scene, float* __restrict__ out)
{
  int gid = blockIdx.x * blockDim.x + threadIdx.x;
  if (gid >= MQ * 2) return;
  int t = gid >> 1, c = gid & 1;
  const float* hr = h2 + (size_t)t * 256;
  const float* wr = w + (size_t)c * 256;
  float acc = bias[c];
#pragma unroll 8
  for (int k = 0; k < 256; ++k) acc += hr[k] * wr[k];
  int s = t & 7, row = t >> 3, n = row & 31, l = row >> 5;
  acc += vin[n * 2 + c] + scene[c];
  out[(((size_t)s * 12 + l) * 32 + n) * 2 + c] = acc;
}

// ============================== launcher ==============================
extern "C" void kernel_launch(void* const* d_in, const int* in_sizes, int n_in,
                              void* d_out, int out_size, void* d_ws, size_t ws_size,
                              hipStream_t stream)
{
  (void)in_sizes; (void)n_in; (void)out_size; (void)ws_size;
  const float* v_in       = (const float*)d_in[0];
  const float* z_in       = (const float*)d_in[1];
  const float* venc_in    = (const float*)d_in[2];
  const float* scene_norm = (const float*)d_in[3];
  const float* input_fc_w = (const float*)d_in[4];
  const float* input_fc_b = (const float*)d_in[5];
  const float* pos_fc_w   = (const float*)d_in[6];
  const float* pos_fc_b   = (const float*)d_in[7];
  const float* sa_ipw     = (const float*)d_in[8];
  const float* sa_ipb     = (const float*)d_in[9];
  const float* sa_ipw_s   = (const float*)d_in[10];
  const float* sa_ipb_s   = (const float*)d_in[11];
  const float* sa_opw     = (const float*)d_in[12];
  const float* sa_opb     = (const float*)d_in[13];
  const float* ca_ipw     = (const float*)d_in[14];
  const float* ca_ipb     = (const float*)d_in[15];
  const float* ca_ipw_s   = (const float*)d_in[16];
  const float* ca_ipb_s   = (const float*)d_in[17];
  const float* ca_opw     = (const float*)d_in[18];
  const float* ca_opb     = (const float*)d_in[19];
  const float* lin1_w     = (const float*)d_in[20];
  const float* lin1_b     = (const float*)d_in[21];
  const float* lin2_w     = (const float*)d_in[22];
  const float* lin2_b     = (const float*)d_in[23];
  const float* n1_g = (const float*)d_in[24];
  const float* n1_b = (const float*)d_in[25];
  const float* n2_g = (const float*)d_in[26];
  const float* n2_b = (const float*)d_in[27];
  const float* n3_g = (const float*)d_in[28];
  const float* n3_b = (const float*)d_in[29];
  const float* mlp1_w   = (const float*)d_in[30];
  const float* mlp1_b   = (const float*)d_in[31];
  const float* mlp2_w   = (const float*)d_in[32];
  const float* mlp2_b   = (const float*)d_in[33];
  const float* out_fc_w = (const float*)d_in[34];
  const float* out_fc_b = (const float*)d_in[35];

  // workspace layout
  char* base = (char*)d_ws;
  size_t off = 0;
  auto take = [&](size_t bytes) -> void* {
    void* p = base + off;
    off = (off + bytes + 255) & ~(size_t)255;
    return p;
  };
  float*    pe     = (float*)take((size_t)12 * E_DIM * 4);
  float*    tgt    = (float*)take((size_t)MQ * E_DIM * 4);
  _Float16* tgt16  = (_Float16*)take((size_t)MQ * E_DIM * 2);
  float*    Hbuf   = (float*)take((size_t)MQ * E_DIM * 4);
  // buf1 region: x1024 f16 (6MB) / qkv f16 (9MB) / ffn-hidden f16 (12MB)
  _Float16* buf1h  = (_Float16*)take((size_t)MQ * DFF_DIM * 2);
  // bufC region: venc-rep f16 (2MB) / qs_ks f16 (6MB) / cross q|qs f16 (6MB)
  _Float16* bufCh  = (_Float16*)take((size_t)MQ * 1024 * 2);
  _Float16* kmem   = (_Float16*)take((size_t)MK * E_DIM * 2);
  _Float16* vmem   = (_Float16*)take((size_t)MK * E_DIM * 2);
  _Float16* ksmem  = (_Float16*)take((size_t)MK * E_DIM * 2);
  float*    scores = (float*)take((size_t)64 * LQ * LQ * 4);
  _Float16* probs  = (_Float16*)take((size_t)64 * LQ * LQ * 2);
  _Float16* attnh  = (_Float16*)take((size_t)MQ * E_DIM * 2);
  _Float16* h1h    = (_Float16*)take((size_t)MQ * E_DIM * 2);
  float*    h2b    = (float*)take((size_t)MQ * 256 * 4);
  const float* nullres = nullptr;
  float*       nullCf  = nullptr;
  _Float16*    nullCh  = nullptr;

#define GEMM(ACT, RES, MODE, Ap, Wp, Bp, Rp, Cfp, Chp, M, N, K) \
  gemm_wmma<ACT, RES, MODE><<<dim3((N) / 128, (M) / 128), 256, 0, stream>>>(Ap, Wp, Bp, Rp, Cfp, Chp, M, N, K)

  // ---- input embedding + positional fc (dual f32+f16 tgt) ----
  pe_kernel<<<(12 * E_DIM + 255) / 256, 256, 0, stream>>>(pe);
  prep_x1024<<<(MQ * E_DIM + 255) / 256, 256, 0, stream>>>(v_in, z_in, input_fc_w, input_fc_b, pe, buf1h);
  GEMM(0, 0, 2, buf1h, pos_fc_w, pos_fc_b, nullres, tgt, tgt16, MQ, E_DIM, 1024);

  // ---- loop-invariant cross-attention K / V / K_s (shared weights, const v_enc) ----
  repeat_venc<<<(MK * E_DIM + 255) / 256, 256, 0, stream>>>(venc_in, bufCh);
  GEMM(0, 0, 1, bufCh, ca_ipw + (size_t)E_DIM * E_DIM,     ca_ipb + E_DIM,     nullres, nullCf, kmem,  MK, E_DIM, E_DIM);
  GEMM(0, 0, 1, bufCh, ca_ipw + (size_t)2 * E_DIM * E_DIM, ca_ipb + 2 * E_DIM, nullres, nullCf, vmem,  MK, E_DIM, E_DIM);
  GEMM(0, 0, 1, bufCh, ca_ipw_s + (size_t)E_DIM * E_DIM,   ca_ipb_s + E_DIM,   nullres, nullCf, ksmem, MK, E_DIM, E_DIM);

  for (int layer = 0; layer < 4; ++layer) {
    // -------- agent-aware self-attention --------
    GEMM(0, 0, 1, tgt16, sa_ipw,   sa_ipb,   nullres, nullCf, buf1h, MQ, 3 * E_DIM, E_DIM);  // q|k|v ld 1536
    GEMM(0, 0, 1, tgt16, sa_ipw_s, sa_ipb_s, nullres, nullCf, bufCh, MQ, 2 * E_DIM, E_DIM);  // qs|ks ld 1024
    attn_scores<1><<<dim3(LQ / 16, LQ / 16, 64), 32, 0, stream>>>(
        buf1h, 3 * E_DIM, buf1h + E_DIM, 3 * E_DIM, bufCh, 2 * E_DIM, bufCh + E_DIM, 2 * E_DIM, scores, LQ);
    softmax_rows<<<64 * LQ, 32, 0, stream>>>(scores, probs, LQ);
    attn_pv<<<dim3(LQ / 16, HD / 16, 64), 32, 0, stream>>>(probs, buf1h + 2 * E_DIM, 3 * E_DIM, attnh, LQ);
    GEMM(0, 1, 0, attnh, sa_opw, sa_opb, tgt, Hbuf, nullCh, MQ, E_DIM, E_DIM);
    layernorm<<<MQ, 32, 0, stream>>>(Hbuf, n1_g, n1_b, tgt, tgt16);

    // -------- agent-aware cross-attention --------
    GEMM(0, 0, 1, tgt16, ca_ipw,   ca_ipb,   nullres, nullCf, bufCh,                      MQ, E_DIM, E_DIM);
    GEMM(0, 0, 1, tgt16, ca_ipw_s, ca_ipb_s, nullres, nullCf, bufCh + (size_t)MQ * E_DIM, MQ, E_DIM, E_DIM);
    attn_scores<0><<<dim3(LQ / 16, LKMEM / 16, 64), 32, 0, stream>>>(
        bufCh, E_DIM, kmem, E_DIM, bufCh + (size_t)MQ * E_DIM, E_DIM, ksmem, E_DIM, scores, LKMEM);
    softmax_rows<<<64 * LQ, 32, 0, stream>>>(scores, probs, LKMEM);
    attn_pv<<<dim3(LQ / 16, HD / 16, 64), 32, 0, stream>>>(probs, vmem, E_DIM, attnh, LKMEM);
    GEMM(0, 1, 0, attnh, ca_opw, ca_opb, tgt, Hbuf, nullCh, MQ, E_DIM, E_DIM);
    layernorm<<<MQ, 32, 0, stream>>>(Hbuf, n2_g, n2_b, tgt, tgt16);

    // -------- FFN --------
    GEMM(1, 0, 1, tgt16, lin1_w, lin1_b, nullres, nullCf, buf1h, MQ, DFF_DIM, E_DIM);
    GEMM(0, 1, 0, buf1h, lin2_w, lin2_b, tgt, Hbuf, nullCh, MQ, E_DIM, DFF_DIM);
    layernorm<<<MQ, 32, 0, stream>>>(Hbuf, n3_g, n3_b, tgt, tgt16);
  }

  // ---- output head ----
  GEMM(1, 0, 1, tgt16, mlp1_w, mlp1_b, nullres, nullCf, h1h, MQ, E_DIM, E_DIM);
  GEMM(1, 0, 0, h1h, mlp2_w, mlp2_b, nullres, h2b, nullCh, MQ, 256, E_DIM);
  head_out<<<(MQ * 2 + 255) / 256, 256, 0, stream>>>(h2b, out_fc_w, out_fc_b, v_in, scene_norm, (float*)d_out);

#undef GEMM
}
